// Cnn_BiLSTM_77369540870598
// MI455X (gfx1250) — matrix operand order repaired and checked by reading, not compiled, not run on hardware
//
#include <hip/hip_runtime.h>
#include <hip/hip_bf16.h>

// ---------------- CDNA5 WMMA types ----------------
typedef __attribute__((ext_vector_type(16))) __bf16 v16bf;
typedef __attribute__((ext_vector_type(8)))  float  v8f;

union AB16 { unsigned u[8]; v16bf v; };

// ---------------- model dims ----------------
constexpr int NPATH = 128;
constexpr int LTOK  = 512;
constexpr int EDIM  = 256;
constexpr int FDIM  = 128;   // conv filters == lstm input
constexpr int CONVT = 510;   // L - K + 1
constexpr int TSEQ  = 509;   // after pool (window 2, stride 1)
constexpr int HDIM  = 128;
constexpr int GDIM  = 512;   // 4*H
constexpr int CKDIM = 768;   // E*K for implicit conv GEMM

// ---------------- helpers ----------------
__device__ __forceinline__ unsigned short f2bf(float f) {
  union { float f; unsigned u; } a; a.f = f;
  unsigned r = a.u + 0x7FFFu + ((a.u >> 16) & 1u);
  return (unsigned short)(r >> 16);
}
__device__ __forceinline__ float bf2f(unsigned short b) {
  union { unsigned u; float f; } a; a.u = ((unsigned)b) << 16;
  return a.f;
}
__device__ __forceinline__ float sigf(float x) { return 1.0f / (1.0f + expf(-x)); }

// =====================================================================
// Kernel 1: weight prep.
//  - Wc2   : conv weights re-ordered bf16, [f][Kg], Kg = k*256 + e  (pairs contiguous)
//  - Wcomb : lstm combined weights bf16, [dir][n][k], k<128 -> w_ih, k>=128 -> w_hh
//  - biasc : b_ih + b_hh per direction, f32
// =====================================================================
__global__ void prep_weights(const float* convw,
                             const float* w_ih_f, const float* w_hh_f,
                             const float* b_ih_f, const float* b_hh_f,
                             const float* w_ih_b, const float* w_hh_b,
                             const float* b_ih_b, const float* b_hh_b,
                             unsigned short* Wc2, unsigned short* Wcomb, float* biasc) {
  int idx = blockIdx.x * blockDim.x + threadIdx.x;
  if (idx < FDIM * CKDIM) {                       // 98304
    int n = idx / CKDIM, Kg = idx % CKDIM;
    int k = Kg >> 8, e = Kg & 255;
    Wc2[idx] = f2bf(convw[n * CKDIM + e * 3 + k]);
  }
  if (idx < 2 * GDIM * 256) {                     // 262144
    int d = idx / (GDIM * 256);
    int n = (idx / 256) % GDIM;
    int k = idx & 255;
    const float* wih = d ? w_ih_b : w_ih_f;
    const float* whh = d ? w_hh_b : w_hh_f;
    float v = (k < FDIM) ? wih[n * FDIM + k] : whh[n * HDIM + (k - FDIM)];
    Wcomb[idx] = f2bf(v);
  }
  if (idx < 2 * GDIM) {
    int d = idx / GDIM, n = idx % GDIM;
    biasc[idx] = d ? (b_ih_b[n] + b_hh_b[n]) : (b_ih_f[n] + b_hh_f[n]);
  }
}

// =====================================================================
// Kernel 2: embedding gather + conv1d as implicit WMMA GEMM + bias + relu.
// Grid: (32 t-tiles, 128 paths), 128 threads (4 waves; each wave -> 32 filters).
// =====================================================================
__global__ __launch_bounds__(128) void conv_kernel(const int* __restrict__ path,
                                                   const float* __restrict__ embA,
                                                   const unsigned short* __restrict__ Wc2,
                                                   const float* __restrict__ convb,
                                                   unsigned short* __restrict__ convbuf) {
  __shared__ unsigned short embt[18 * EDIM];      // 18 token rows (16 + K-1 halo), bf16
  const int p  = blockIdx.y;
  const int t0 = blockIdx.x * 16;
  const int tid = threadIdx.x;

  for (int i = tid; i < 18 * EDIM; i += 128) {
    int r = i >> 8, e = i & 255;
    int tok_t = t0 + r; if (tok_t > LTOK - 1) tok_t = LTOK - 1;
    int tok = path[p * LTOK + tok_t];
    embt[i] = f2bf(embA[(long)tok * EDIM + e]);
  }
  __syncthreads();

  const int lane = tid & 31, wid = tid >> 5;
  const int hi = (lane >> 4) & 1;
  const int mrow = lane & 15;
  const int ncol = lane & 15;
  const unsigned* Wc2u = (const unsigned*)Wc2;

  v8f acc[2];
  #pragma unroll
  for (int i = 0; i < 8; ++i) { acc[0][i] = 0.f; acc[1][i] = 0.f; }

  for (int ks = 0; ks < 24; ++ks) {               // K = 768 = 24 * 32
    AB16 a;
    #pragma unroll
    for (int v = 0; v < 8; ++v) {                 // A layout: 16-bit 16x32 (ISA 7.12.2)
      int kl = ((v >> 2) << 4) + (hi << 3) + ((v & 3) << 1);
      int Kg = ks * 32 + kl;
      int k = Kg >> 8, e = Kg & 255;              // pair (e, e+1), same k
      a.u[v] = *(const unsigned*)&embt[(mrow + k) * EDIM + e];
    }
    #pragma unroll
    for (int nn = 0; nn < 2; ++nn) {
      int n = (wid * 2 + nn) * 16 + ncol;
      AB16 b;
      #pragma unroll
      for (int v = 0; v < 8; ++v) {               // B layout: K pair = (v>>2)*16 + hi*8 + 2*(v&3)
        int kk = ks * 32 + ((v >> 2) << 4) + (hi << 3) + ((v & 3) << 1);
        b.u[v] = Wc2u[(n * CKDIM + kk) >> 1];
      }
      acc[nn] = __builtin_amdgcn_wmma_f32_16x16x32_bf16(
          false, a.v, false, b.v, (short)0, acc[nn], false, false);
    }
  }
  #pragma unroll
  for (int nn = 0; nn < 2; ++nn) {
    int n = (wid * 2 + nn) * 16 + ncol;
    float bias = convb[n];
    #pragma unroll
    for (int v = 0; v < 8; ++v) {
      int m = hi * 8 + v;
      int t = t0 + m;
      if (t < CONVT) {
        float r = fmaxf(acc[nn][v] + bias, 0.f);
        convbuf[((long)p * CONVT + t) * FDIM + n] = f2bf(r);
      }
    }
  }
}

// =====================================================================
// Kernel 3: max-pool window 2, stride 1 -> pooled [P][TSEQ][F] bf16
// =====================================================================
__global__ void pool_kernel(const unsigned short* __restrict__ convbuf,
                            unsigned short* __restrict__ pooled) {
  long idx = (long)blockIdx.x * blockDim.x + threadIdx.x;
  const long N = (long)NPATH * TSEQ * FDIM;
  if (idx >= N) return;
  int f = (int)(idx % FDIM);
  int t = (int)((idx / FDIM) % TSEQ);
  int p = (int)(idx / ((long)FDIM * TSEQ));
  float a = bf2f(convbuf[((long)p * CONVT + t) * FDIM + f]);
  float b = bf2f(convbuf[((long)p * CONVT + t + 1) * FDIM + f]);
  pooled[idx] = f2bf(fmaxf(a, b));
}

// =====================================================================
// Kernel 4: persistent BiLSTM. 16 blocks = {2 dirs} x {8 blocks of 16 paths},
// 256 threads (8 waves; each wave -> 64 gate columns).
// Combined weights stay resident in VGPRs (4 nt x 8 ks x 8 = 256 regs/wave).
// =====================================================================
__global__ __launch_bounds__(256) void lstm_kernel(const unsigned short* __restrict__ pooled,
                                                   const unsigned short* __restrict__ Wcomb,
                                                   const float* __restrict__ biasc,
                                                   float* __restrict__ context) {
  __shared__ float gates[16 * GDIM];              // 32 KB gate exchange
  __shared__ float cst[16 * HDIM];                //  8 KB cell state
  __shared__ unsigned short hst[16 * HDIM];       //  4 KB hidden (bf16, feeds A frags)

  const int tid = threadIdx.x;
  const int lane = tid & 31, wid = tid >> 5;
  const int hi = (lane >> 4) & 1;
  const int mrow = lane & 15, ncol = lane & 15;
  const int d  = blockIdx.x >> 3;                 // direction
  const int p0 = (blockIdx.x & 7) * 16;           // first path of this block

  for (int i = tid; i < 16 * HDIM; i += 256) { cst[i] = 0.f; hst[i] = 0; }

  // Preload stationary B fragments + per-column bias.
  unsigned Bfrag[4][8][8];
  float bias_nt[4];
  const unsigned* Wu = (const unsigned*)Wcomb;
  #pragma unroll
  for (int nt = 0; nt < 4; ++nt) {
    int n = (wid * 4 + nt) * 16 + ncol;
    bias_nt[nt] = biasc[d * GDIM + n];
    #pragma unroll
    for (int ks = 0; ks < 8; ++ks) {
      #pragma unroll
      for (int v = 0; v < 8; ++v) {
        int kk = ks * 32 + ((v >> 2) << 4) + (hi << 3) + ((v & 3) << 1);
        Bfrag[nt][ks][v] = Wu[((long)(d * GDIM + n)) * 128 + (kk >> 1)];
      }
    }
  }
  __syncthreads();

  for (int t = 0; t < TSEQ; ++t) {
    const int tt = d ? (TSEQ - 1 - t) : t;
    v8f acc[4];
    #pragma unroll
    for (int nt = 0; nt < 4; ++nt)
      #pragma unroll
      for (int i = 0; i < 8; ++i) acc[nt][i] = bias_nt[nt];

    #pragma unroll
    for (int ks = 0; ks < 8; ++ks) {              // K = 256: ks<4 -> x_t, ks>=4 -> h
      AB16 a;
      if (ks < 4) {
        const unsigned* xr = (const unsigned*)pooled + ((long)(p0 + mrow) * TSEQ + tt) * 64;
        #pragma unroll
        for (int v = 0; v < 8; ++v) {
          int kl = ((v >> 2) << 4) + (hi << 3) + ((v & 3) << 1);
          a.u[v] = xr[(ks * 32 + kl) >> 1];
        }
      } else {
        #pragma unroll
        for (int v = 0; v < 8; ++v) {
          int kl = ((v >> 2) << 4) + (hi << 3) + ((v & 3) << 1);
          int kk = (ks - 4) * 32 + kl;
          a.u[v] = *(const unsigned*)&hst[mrow * HDIM + kk];
        }
      }
      #pragma unroll
      for (int nt = 0; nt < 4; ++nt) {
        AB16 b;
        #pragma unroll
        for (int v2 = 0; v2 < 8; ++v2) b.u[v2] = Bfrag[nt][ks][v2];
        acc[nt] = __builtin_amdgcn_wmma_f32_16x16x32_bf16(
            false, a.v, false, b.v, (short)0, acc[nt], false, false);
      }
    }

    #pragma unroll
    for (int nt = 0; nt < 4; ++nt) {
      int n = (wid * 4 + nt) * 16 + ncol;
      #pragma unroll
      for (int v = 0; v < 8; ++v) gates[(hi * 8 + v) * GDIM + n] = acc[nt][v];
    }
    __syncthreads();

    #pragma unroll
    for (int r = 0; r < 8; ++r) {                 // gate math: i,f,g,o (torch order)
      int idx = tid * 8 + r;
      int m = idx >> 7, j = idx & 127;
      float gi = gates[m * GDIM + j];
      float gf = gates[m * GDIM + 128 + j];
      float gg = gates[m * GDIM + 256 + j];
      float go = gates[m * GDIM + 384 + j];
      float c = cst[m * HDIM + j];
      c = sigf(gf) * c + sigf(gi) * tanhf(gg);
      cst[m * HDIM + j] = c;
      hst[m * HDIM + j] = f2bf(sigf(go) * tanhf(c));
    }
    __syncthreads();
  }

  for (int i = tid; i < 16 * HDIM; i += 256) {
    int m = i >> 7, j = i & 127;
    context[(p0 + m) * EDIM + d * HDIM + j] = bf2f(hst[i]);
  }
}

// =====================================================================
// Kernel 5: 2-step additive attention pooling + final MLP (tiny, f32 VALU).
// =====================================================================
__global__ __launch_bounds__(256) void attn_kernel(const float* __restrict__ ctx,
                                                   const float* __restrict__ embB,
                                                   const int* __restrict__ queryp,
                                                   const float* __restrict__ d1w,
                                                   const float* __restrict__ d1b,
                                                   const float* __restrict__ d2w,
                                                   const float* __restrict__ d2b,
                                                   float* __restrict__ out) {
  __shared__ float uvec[EDIM], ovec[EDIM], alpha[NPATH], part[256], unew[EDIM];
  const int tid = threadIdx.x;
  uvec[tid] = embB[(long)(*queryp) * EDIM + tid];
  __syncthreads();

  for (int iter = 0; iter < 2; ++iter) {
    int p = tid >> 1, half = tid & 1;
    float s = 0.f;
    for (int j = half * 128; j < half * 128 + 128; ++j) {
      float a = d1b[j];
      const float* wr = d1w + j * 512;
      for (int k = 0; k < 256; ++k) a += ctx[p * EDIM + k] * wr[k];
      for (int k = 0; k < 256; ++k) a += uvec[k] * wr[256 + k];
      s += tanhf(a) * d2w[j];
    }
    part[tid] = s;
    __syncthreads();
    if (tid < NPATH) alpha[tid] = part[2 * tid] + part[2 * tid + 1] + d2b[0];
    __syncthreads();
    if (tid == 0) {
      float mx = alpha[0];
      for (int q = 1; q < NPATH; ++q) mx = fmaxf(mx, alpha[q]);
      float sm = 0.f;
      for (int q = 0; q < NPATH; ++q) { float e = expf(alpha[q] - mx); alpha[q] = e; sm += e; }
      float inv = 1.f / sm;
      for (int q = 0; q < NPATH; ++q) alpha[q] *= inv;
    }
    __syncthreads();
    float oe = 0.f;
    for (int q = 0; q < NPATH; ++q) oe += alpha[q] * ctx[q * EDIM + tid];
    ovec[tid] = oe;
    __syncthreads();
    {
      float a = d1b[tid];
      const float* wr = d1w + tid * 512;
      for (int k = 0; k < 256; ++k) a += uvec[k] * wr[k];
      for (int k = 0; k < 256; ++k) a += ovec[k] * wr[256 + k];
      unew[tid] = a;
    }
    __syncthreads();
    uvec[tid] = unew[tid];
    __syncthreads();
  }

  part[tid] = fmaxf(uvec[tid], 0.f) * d2w[tid];
  __syncthreads();
  if (tid == 0) {
    float s = d2b[0];
    for (int k = 0; k < 256; ++k) s += part[k];
    out[0] = 1.f / (1.f + expf(-s));
  }
}

// =====================================================================
extern "C" void kernel_launch(void* const* d_in, const int* in_sizes, int n_in,
                              void* d_out, int out_size, void* d_ws, size_t ws_size,
                              hipStream_t stream) {
  const int*   path   = (const int*)  d_in[0];
  const int*   query  = (const int*)  d_in[1];
  const float* embA   = (const float*)d_in[2];
  const float* embB   = (const float*)d_in[3];
  const float* convw  = (const float*)d_in[4];
  const float* convb  = (const float*)d_in[5];
  const float* w_ih_f = (const float*)d_in[6];
  const float* w_hh_f = (const float*)d_in[7];
  const float* b_ih_f = (const float*)d_in[8];
  const float* b_hh_f = (const float*)d_in[9];
  const float* w_ih_b = (const float*)d_in[10];
  const float* w_hh_b = (const float*)d_in[11];
  const float* b_ih_b = (const float*)d_in[12];
  const float* b_hh_b = (const float*)d_in[13];
  const float* d1w    = (const float*)d_in[14];
  const float* d1b    = (const float*)d_in[15];
  const float* d2w    = (const float*)d_in[16];
  const float* d2b    = (const float*)d_in[17];
  float* out = (float*)d_out;

  char* ws = (char*)d_ws;
  size_t off = 0;
  auto alloc = [&](size_t bytes) -> void* {
    void* p = ws + off;
    off = (off + bytes + 255) & ~(size_t)255;
    return p;
  };
  unsigned short* convbuf = (unsigned short*)alloc((size_t)NPATH * CONVT * FDIM * 2); // 16.7 MB
  unsigned short* pooled  = (unsigned short*)alloc((size_t)NPATH * TSEQ * FDIM * 2);  // 16.7 MB
  unsigned short* Wc2     = (unsigned short*)alloc((size_t)FDIM * CKDIM * 2);         // 196 KB
  unsigned short* Wcomb   = (unsigned short*)alloc((size_t)2 * GDIM * 256 * 2);       // 512 KB
  float*          biasc   = (float*)alloc((size_t)2 * GDIM * 4);
  float*          context = (float*)alloc((size_t)NPATH * EDIM * 4);

  prep_weights<<<dim3(1024), dim3(256), 0, stream>>>(
      convw, w_ih_f, w_hh_f, b_ih_f, b_hh_f,
      w_ih_b, w_hh_b, b_ih_b, b_hh_b, Wc2, Wcomb, biasc);

  conv_kernel<<<dim3(32, 128), dim3(128), 0, stream>>>(path, embA, Wc2, convb, convbuf);

  {
    long N = (long)NPATH * TSEQ * FDIM;
    int blocks = (int)((N + 255) / 256);
    pool_kernel<<<dim3(blocks), dim3(256), 0, stream>>>(convbuf, pooled);
  }

  lstm_kernel<<<dim3(16), dim3(256), 0, stream>>>(pooled, Wcomb, biasc, context);

  attn_kernel<<<dim3(1), dim3(256), 0, stream>>>(context, embB, query, d1w, d1b, d2w, d2b, out);
}
